// GraphConvolution_60748017435371
// MI455X (gfx1250) — matrix operand order repaired and checked by reading, not compiled, not run on hardware
//
#include <hip/hip_runtime.h>

// ---------------------------------------------------------------------------
// GCN layer for MI455X (gfx1250, wave32):
//   support = x @ W + b        (BF16 WMMA 16x16x32, f32 accumulate)
//   out     = scatter_add(support[cols] * vals, rows)   (f32 atomics, L2 gather)
// ---------------------------------------------------------------------------

typedef __attribute__((ext_vector_type(16))) __bf16 v16bf;
typedef __attribute__((ext_vector_type(8)))  float  v8f;

#define DDIM 128            // D_IN == D_OUT == 128
#define LDS_STRIDE 136      // padded row stride in halves: 272 B = 17*16 B
                            //  -> every 16-half sub-run is 16B aligned,
                            //  -> rows shift 4 banks apiece (conflict-free frags)

__device__ __forceinline__ unsigned short f2bf(float f) {
    // round-to-nearest-even fp32 -> bf16
    unsigned u = __builtin_bit_cast(unsigned, f);
    return (unsigned short)((u + 0x7FFFu + ((u >> 16) & 1u)) >> 16);
}

__device__ __forceinline__ unsigned pack2bf(float lo, float hi) {
    return (unsigned)f2bf(lo) | ((unsigned)f2bf(hi) << 16);
}

union Frag {
    unsigned short u[16];
    uint4          q[2];
    v16bf          v;
};

// ---------------------------------------------------------------------------
// Kernel 1: support = x @ W + b
// 256 threads = 8 waves; wave w owns output cols [16w,16w+16) of a 16-row
// tile. Grid-stride over row tiles; W staged (transposed, bf16) in LDS once
// per block. K-loop: 4x v_wmma_f32_16x16x32_bf16, fragments via ds_load_b128.
// ---------------------------------------------------------------------------
__global__ void __launch_bounds__(256)
gcn_gemm_wmma(const float* __restrict__ x,
              const float* __restrict__ W,
              const float* __restrict__ b,
              float* __restrict__ sup,
              int nodes, int rowTiles) {
    __shared__ unsigned short ldsWT[DDIM * LDS_STRIDE];  // W^T as bf16: [n][k]
    __shared__ unsigned short ldsX [16   * LDS_STRIDE];  // x tile as bf16: [m][k]
    __shared__ float          ldsB [DDIM];

    const int tid   = threadIdx.x;
    const int wave  = tid >> 5;
    const int lane  = tid & 31;
    const int lhalf = lane >> 4;     // 0: lanes 0-15, 1: lanes 16-31
    const int lm    = lane & 15;

    // Stage W transposed (fp32 -> bf16, packed b32 LDS stores) + bias.
    // idx -> (k2, n): global reads coalesced along n; k pair contiguous in LDS.
    for (int idx = tid; idx < (DDIM / 2) * DDIM; idx += 256) {
        const int k2 = idx >> 7;            // k pair index 0..63
        const int n  = idx & (DDIM - 1);
        const float w0 = W[(size_t)(2 * k2 + 0) * DDIM + n];
        const float w1 = W[(size_t)(2 * k2 + 1) * DDIM + n];
        *(unsigned*)&ldsWT[n * LDS_STRIDE + 2 * k2] = pack2bf(w0, w1);
    }
    if (tid < DDIM) ldsB[tid] = b[tid];
    __syncthreads();

    const int   ncol = wave * 16 + lm;      // this lane's output column
    const float bias = ldsB[ncol];

    // Per-thread staging coords for the x tile (4 pairs/thread).
    const int sr = tid >> 6;                // base row 0..3 (step 4)
    const int sc = (tid & 63) * 2;          // even column

    for (int rt = blockIdx.x; rt < rowTiles; rt += gridDim.x) {
        const int  rowbase = rt * 16;
        const bool full    = (rowbase + 16 <= nodes);   // tile fully in range

        __syncthreads();   // previous tile fully consumed before overwrite
        // Stage x tile (16 x 128) as bf16, packed b32 stores.
        if (full) {
            const float* p = x + (size_t)(rowbase + sr) * DDIM + sc;
            #pragma unroll
            for (int rr = 0; rr < 4; ++rr) {
                const float* q = p + (size_t)(rr * 4) * DDIM;
                *(unsigned*)&ldsX[(sr + rr * 4) * LDS_STRIDE + sc] =
                    pack2bf(q[0], q[1]);
            }
        } else {
            #pragma unroll
            for (int rr = 0; rr < 4; ++rr) {
                const int r   = sr + rr * 4;
                const int row = rowbase + r;
                float x0 = 0.f, x1 = 0.f;
                if (row < nodes) {
                    const float* p = x + (size_t)row * DDIM + sc;
                    x0 = p[0]; x1 = p[1];
                }
                *(unsigned*)&ldsX[r * LDS_STRIDE + sc] = pack2bf(x0, x1);
            }
        }
        __syncthreads();

        v8f acc = {};
        #pragma unroll
        for (int k0 = 0; k0 < DDIM; k0 += 32) {
            Frag A, B;
            // A 16x32 bf16 (ISA 7.12.2): lane (lhalf,lm) holds row lm,
            // K = k0 + 8*lhalf + {0..7} and {16..23}  -> two aligned b128s.
            const unsigned short* pa = &ldsX[lm * LDS_STRIDE + k0 + lhalf * 8];
            A.q[0] = *(const uint4*)(pa);
            A.q[1] = *(const uint4*)(pa + 16);
            // B 32x16 bf16: lane holds col ncol, K = k0 + 8*lhalf + {0..7} and {16..23}
            // -> transposed LDS: two aligned b128s.
            const unsigned short* pb = &ldsWT[ncol * LDS_STRIDE + k0 + lhalf * 8];
            B.q[0] = *(const uint4*)(pb);
            B.q[1] = *(const uint4*)(pb + 16);

            acc = __builtin_amdgcn_wmma_f32_16x16x32_bf16(
                /*neg_a=*/false, A.v, /*neg_b=*/false, B.v,
                /*c_mod=*/(short)0, acc, /*reuse_a=*/false, /*reuse_b=*/false);
        }

        // C/D layout: VGPR i -> M = i + 8*lhalf, N = lm (wave's 16-col slice).
        if (full) {
            // Single base address; 8 stores with immediate offsets (i*512B).
            float* p = sup + (size_t)(rowbase + lhalf * 8) * DDIM + ncol;
            #pragma unroll
            for (int i = 0; i < 8; ++i)
                p[(size_t)i * DDIM] = acc[i] + bias;
        } else {
            #pragma unroll
            for (int i = 0; i < 8; ++i) {
                const int row = rowbase + lhalf * 8 + i;
                if (row < nodes)
                    sup[(size_t)row * DDIM + ncol] = acc[i] + bias;
            }
        }
    }
}

// ---------------------------------------------------------------------------
// Kernel 2: zero the output (harness poisons d_out).
// ---------------------------------------------------------------------------
__global__ void zero_out_kernel(float4* __restrict__ out, int n4) {
    const int i = blockIdx.x * blockDim.x + threadIdx.x;
    if (i < n4) out[i] = make_float4(0.f, 0.f, 0.f, 0.f);
}

// ---------------------------------------------------------------------------
// Kernel 3: out[rows[e]] += support[cols[e]] * vals[e]
// One 32-lane wave per edge. Edge index is forced into an SGPR via
// readfirstlane so rows/cols/vals become scalar (SMEM) loads; each lane does
// one B128 gather + 4 global_atomic_add_f32. support (51 MB) lives in L2.
// ---------------------------------------------------------------------------
__global__ void __launch_bounds__(256)
gcn_scatter(const float* __restrict__ sup,
            const int*   __restrict__ rows,
            const int*   __restrict__ cols,
            const float* __restrict__ vals,
            float* __restrict__ out,
            int nEdges) {
    // wave-uniform edge index, forced scalar
    const int e = __builtin_amdgcn_readfirstlane(
        blockIdx.x * (blockDim.x >> 5) + (threadIdx.x >> 5));
    if (e >= nEdges) return;
    const int lane = threadIdx.x & 31;

    const int   r = rows[e];   // s_load (uniform)
    const int   c = cols[e];   // s_load (uniform)
    const float v = vals[e];   // s_load (uniform)

    const float4 s = ((const float4*)(sup + (size_t)c * DDIM))[lane];
    float* dst = out + (size_t)r * DDIM + lane * 4;
    atomicAdd(dst + 0, s.x * v);
    atomicAdd(dst + 1, s.y * v);
    atomicAdd(dst + 2, s.z * v);
    atomicAdd(dst + 3, s.w * v);
}

// ---------------------------------------------------------------------------
extern "C" void kernel_launch(void* const* d_in, const int* in_sizes, int n_in,
                              void* d_out, int out_size, void* d_ws, size_t ws_size,
                              hipStream_t stream) {
    const float* x    = (const float*)d_in[0];
    const int*   rows = (const int*)  d_in[1];
    const int*   cols = (const int*)  d_in[2];
    const float* vals = (const float*)d_in[3];
    const float* W    = (const float*)d_in[4];
    const float* b    = (const float*)d_in[5];
    float*       out  = (float*)d_out;
    float*       sup  = (float*)d_ws;          // needs nodes*128*4 bytes (51.2 MB)

    const int nodes    = in_sizes[0] / DDIM;
    const int nEdges   = in_sizes[1];
    const int rowTiles = (nodes + 15) / 16;

    // 1) GEMM (persistent blocks, W staged once per block)
    const int gemmBlocks = rowTiles < 1024 ? rowTiles : 1024;
    gcn_gemm_wmma<<<gemmBlocks, 256, 0, stream>>>(x, W, b, sup, nodes, rowTiles);

    // 2) zero output
    const int n4 = out_size / 4;
    zero_out_kernel<<<(n4 + 255) / 256, 256, 0, stream>>>((float4*)out, n4);

    // 3) edge scatter-add: one wave (32 lanes) per edge, 8 edges per block
    const int sblocks = (nEdges + 7) / 8;
    gcn_scatter<<<sblocks, 256, 0, stream>>>(sup, rows, cols, vals, out, nEdges);
}
